// AttentionOnDetail_72619307041425
// MI455X (gfx1250) — compile-verified, matrix operand order checked by reading, not run on hardware
//
#include <hip/hip_runtime.h>
#include <hip/hip_bf16.h>
#include <math.h>

// ---------------------------------------------------------------------------
// Types / WMMA helpers for gfx1250 (CDNA5, wave32)
// ---------------------------------------------------------------------------
typedef __bf16 bf16;
typedef __attribute__((ext_vector_type(16))) __bf16 v16bf;
typedef __attribute__((ext_vector_type(8)))  __bf16 v8bf;
typedef __attribute__((ext_vector_type(8)))  float  v8f;

__device__ __forceinline__ int imin(int a, int b) { return a < b ? a : b; }

__device__ __forceinline__ v8f wmma_bf16(v16bf a, v16bf b, v8f c) {
  // D(f32 16x16) = A(bf16 16x32) * B(bf16 32x16) + C
  return __builtin_amdgcn_wmma_f32_16x16x32_bf16(
      /*neg_a=*/false, a, /*neg_b=*/false, b,
      /*c_mod=*/(short)0, c, /*reuse_a=*/false, /*reuse_b=*/false);
}

// Scheduler pipelining directives (keep VMEM reads interleaved ahead of WMMA).
// mask: 0x008 = MFMA/WMMA group, 0x020 = VMEM read group.
#if __has_builtin(__builtin_amdgcn_sched_group_barrier)
#define SG(mask, size) __builtin_amdgcn_sched_group_barrier((mask), (size), 0)
#else
#define SG(mask, size)
#endif

// 16-bit A/B fragment where the per-lane K-run is CONTIGUOUS in memory:
//   element e (0..15): group = e>>3, i = e&7, k = koff + group*16 + half*8 + i
// i.e. two contiguous 8-element (16-byte) chunks -> two b128 loads.
__device__ __forceinline__ v16bf frag_ld_contig(const bf16* p, int half) {
  const v8bf lo = *(const v8bf*)(p + half * 8);
  const v8bf hi = *(const v8bf*)(p + 16 + half * 8);
  v16bf f;
#pragma unroll
  for (int i = 0; i < 8; ++i) { f[i] = lo[i]; f[8 + i] = hi[i]; }
  return f;
}

__device__ __forceinline__ v8f zero8() {
  v8f z = {0.f, 0.f, 0.f, 0.f, 0.f, 0.f, 0.f, 0.f};
  return z;
}

// ---------------------------------------------------------------------------
// Problem constants (from the reference)
// ---------------------------------------------------------------------------
#define BATCH 4
#define TSEQ  2048
#define T1    2049          // with sink token
#define CEMB  256
#define NHEAD 8
#define HDIM  256           // head dim == n_embd
#define NQKV  2048          // NHEAD * HDIM (the y row width for out-proj)
#define VTS   2080          // padded t-stride of transposed V/G (mult of 8, >= 2048+32)
#define RMS_EPS 1.1920929e-7f

// ---------------------------------------------------------------------------
// Prep kernels (bandwidth-bound)
// ---------------------------------------------------------------------------
__global__ void cvt_f32_bf16(const float* __restrict__ src, bf16* __restrict__ dst, int n) {
  int i = blockIdx.x * 256 + threadIdx.x;
  if (i < n) dst[i] = (bf16)src[i];
}

__global__ void zero_bf16(bf16* __restrict__ dst, int n) {
  int i = blockIdx.x * 256 + threadIdx.x;
  if (i < n) dst[i] = (bf16)0.0f;
}

__global__ void build_x0(const float* __restrict__ x, const float* __restrict__ wsink,
                         bf16* __restrict__ x0) {
  int i = blockIdx.x * 256 + threadIdx.x;
  if (i >= BATCH * T1 * CEMB) return;
  int c = i & (CEMB - 1);
  int t = (i >> 8) % T1;
  int b = i / (T1 * CEMB);
  float v = (t == 0) ? wsink[c] : x[((size_t)b * TSEQ + (t - 1)) * CEMB + c];
  x0[i] = (bf16)v;
}

// ---------------------------------------------------------------------------
// Fused QKVG projection + RoPE + RMSNorm
// grid: (129 token-tiles, B*H, 4 [0=q,1=k,2=v,3=g]), block: 64 (2 waves)
// Each block: 16x256 bf16 GEMM tile (K=256) via WMMA, then f32 postprocess.
// V and G are stored TRANSPOSED: [bh][d][t] with t-stride VTS so the
// attention P*V B-operand and sigmoid-gate reads become b128 loads.
// ---------------------------------------------------------------------------
__global__ void __launch_bounds__(64)
qkvg_kernel(const bf16* __restrict__ X0, const bf16* __restrict__ Wb,
            const float* __restrict__ cosp, const float* __restrict__ sinp,
            const float* __restrict__ tao,
            bf16* __restrict__ Qo, bf16* __restrict__ Ko,
            bf16* __restrict__ Vt, bf16* __restrict__ Gt) {
  const int tid  = threadIdx.x;
  const int lane = tid & 31;
  const int wave = tid >> 5;          // 0..1, each owns 128 output cols
  const int half = lane >> 4;
  const int ln   = lane & 15;

  const int qt0 = blockIdx.x * 16;    // token-tile base (0..2048, tiles of 16)
  const int b   = blockIdx.y >> 3;
  const int h   = blockIdx.y & 7;
  const int s   = blockIdx.z;         // 0=q 1=k 2=v 3=g

  __shared__ float Lf[16 * 256];
  __shared__ float rowscale[16];

  // ---- 16x256 GEMM tile, K = 256, pipelined B stream ----
  v8f acc[8];
#pragma unroll
  for (int t2 = 0; t2 < 8; ++t2) acc[t2] = zero8();

  const int trow = imin(qt0 + ln, TSEQ);                 // clamp OOB rows
  const bf16* xr = X0 + ((size_t)b * T1 + trow) * CEMB;  // A row
  const int colbase = h * 1024 + s * 256 + wave * 128;   // W_qkvg row index base
  const bf16* wr0 = Wb + (size_t)(colbase + ln) * CEMB;  // B row for tile 0

#pragma unroll
  for (int kk = 0; kk < 8; ++kk) {                       // K loop: 8 x 32
    v16bf a = frag_ld_contig(xr + kk * 32, half);
    v16bf bcur = frag_ld_contig(wr0 + kk * 32, half);
#pragma unroll
    for (int t2 = 0; t2 < 8; ++t2) {
      v16bf bnxt = bcur;
      if (t2 < 7)
        bnxt = frag_ld_contig(Wb + (size_t)(colbase + (t2 + 1) * 16 + ln) * CEMB + kk * 32, half);
      acc[t2] = wmma_bf16(a, bcur, acc[t2]);
      bcur = bnxt;
      SG(0x020, 4);   // keep ~4 VMEM reads in flight ...
      SG(0x008, 1);   // ... per WMMA
    }
  }

  // C layout -> LDS f32 tile
#pragma unroll
  for (int t2 = 0; t2 < 8; ++t2)
#pragma unroll
    for (int j = 0; j < 8; ++j)
      Lf[(j + 8 * half) * 256 + wave * 128 + t2 * 16 + ln] = acc[t2][j];
  __syncthreads();

  // ---- RoPE + RMSNorm (q,k only) ----
  if (s < 2) {
    for (int idx = tid; idx < 16 * 128; idx += 64) {
      int r = idx >> 7, d = idx & 127;
      int t  = imin(qt0 + r, TSEQ);
      float c  = cosp[t * 128 + d];
      float sn = sinp[t * 128 + d];
      float x1 = Lf[r * 256 + d];
      float x2 = Lf[r * 256 + 128 + d];
      Lf[r * 256 + d]       =  x1 * c + x2 * sn;
      Lf[r * 256 + 128 + d] = -x1 * sn + x2 * c;
    }
    __syncthreads();
    if (tid < 16) {
      float ss = 0.f;
      for (int d = 0; d < 256; ++d) { float v = Lf[tid * 256 + d]; ss += v * v; }
      rowscale[tid] = rsqrtf(ss * (1.0f / 256.0f) + RMS_EPS) * tao[s];
    }
    __syncthreads();
  }

  // ---- store bf16 ----
  const int bh = b * NHEAD + h;
  if (s >= 2) {
    // transposed store: [bh][d][t], r fastest -> coalesced runs along t
    bf16* dst = (s == 2) ? Vt : Gt;
    for (int idx = tid; idx < 16 * 256; idx += 64) {
      int d = idx >> 4, r = idx & 15;
      int t = qt0 + r;
      float v = (t <= TSEQ) ? Lf[r * 256 + d] : 0.0f;   // zero the row tail
      dst[((size_t)bh * HDIM + d) * VTS + t] = (bf16)v;
    }
  } else {
    bf16* dst = (s == 0) ? Qo : Ko;
    for (int idx = tid; idx < 16 * 256; idx += 64) {
      int r = idx >> 8, d = idx & 255;
      int t = qt0 + r;
      if (t <= TSEQ) {
        float v = Lf[r * 256 + d] * rowscale[r];
        dst[((size_t)bh * T1 + t) * HDIM + d] = (bf16)v;
      }
    }
  }
}

// ---------------------------------------------------------------------------
// Flash attention (causal, with sink row 0).
// grid: (33, B*H), block: 128 = 4 independent waves, one 16-query tile each.
// S = Q K^T and O += P V both via v_wmma_f32_16x16x32_bf16 with scheduler
// pipelining hints; P transposed C->A layout through a private 1KB LDS slice
// (in-order DS within a wave, no barriers).
// ---------------------------------------------------------------------------
__global__ void __launch_bounds__(128)
attn_kernel(const bf16* __restrict__ Q, const bf16* __restrict__ K,
            const bf16* __restrict__ Vt, const bf16* __restrict__ Gt,
            bf16* __restrict__ Y) {
  const int lane = threadIdx.x & 31;
  const int wave = threadIdx.x >> 5;
  const int half = lane >> 4;
  const int ln   = lane & 15;
  const int tile = blockIdx.x * 4 + wave;
  if (tile > 128) return;                    // 129 tiles total
  const int qt0  = tile * 16;
  const int bh   = blockIdx.y;
  const int b    = bh >> 3;
  const int h    = bh & 7;

  const bf16* Qb  = Q  + (size_t)bh * T1 * HDIM;
  const bf16* Kb  = K  + (size_t)bh * T1 * HDIM;
  const bf16* Vtb = Vt + (size_t)bh * HDIM * VTS;
  const bf16* Gtb = Gt + (size_t)bh * HDIM * VTS;

  __shared__ bf16 Pl[4 * 16 * 32];           // per-wave P staging
  bf16* Pw = Pl + wave * (16 * 32);

  float m8[8], l8[8];
  v8f O[16];
#pragma unroll
  for (int j = 0; j < 8; ++j) { m8[j] = -3.0e38f; l8[j] = 0.f; }
#pragma unroll
  for (int t = 0; t < 16; ++t) O[t] = zero8();

  const int qmax = imin(qt0 + 15, TSEQ);     // causal range for this tile
  const int qrow = imin(qt0 + ln, TSEQ);
  const bf16* qp = Qb + (size_t)qrow * HDIM; // A rows (lane = M)

  for (int kb = 0; kb <= qmax; kb += 32) {   // stream 32 keys per step
    // prefetch next chunk's K rows (global_prefetch_b8)
    if (kb + 32 <= qmax) {
      __builtin_prefetch((const void*)(Kb + (size_t)imin(kb + 32 + ln, TSEQ) * HDIM), 0, 1);
      __builtin_prefetch((const void*)(Kb + (size_t)imin(kb + 48 + ln, TSEQ) * HDIM), 0, 1);
    }

    // ---- S = Q K^T (two 16x16 tiles, K-dim 256 = 8 WMMAs each) ----
    v8f S0 = zero8(), S1 = zero8();
    const int key0c = imin(kb + ln, TSEQ);
    const int key1c = imin(kb + 16 + ln, TSEQ);
    const bf16* kp0 = Kb + (size_t)key0c * HDIM;
    const bf16* kp1 = Kb + (size_t)key1c * HDIM;

    v16bf a_c  = frag_ld_contig(qp,  half);
    v16bf b0_c = frag_ld_contig(kp0, half);
    v16bf b1_c = frag_ld_contig(kp1, half);
#pragma unroll
    for (int kk = 0; kk < 8; ++kk) {
      v16bf a_n = a_c, b0_n = b0_c, b1_n = b1_c;
      if (kk < 7) {
        a_n  = frag_ld_contig(qp  + (kk + 1) * 32, half);
        b0_n = frag_ld_contig(kp0 + (kk + 1) * 32, half);
        b1_n = frag_ld_contig(kp1 + (kk + 1) * 32, half);
      }
      S0 = wmma_bf16(a_c, b0_c, S0);
      S1 = wmma_bf16(a_c, b1_c, S1);
      a_c = a_n; b0_c = b0_n; b1_c = b1_n;
      SG(0x020, 3);   // 6 VMEM reads / 2 WMMAs per kk, interleaved
      SG(0x008, 1);
      SG(0x020, 3);
      SG(0x008, 1);
    }

    // ---- scale, causal mask, online softmax (row = j + 8*half, col = ln) ----
    float alpha[8];
#pragma unroll
    for (int j = 0; j < 8; ++j) {
      const int q_t = qt0 + j + 8 * half;
      float s0 = S0[j] * 0.0625f;            // 1/sqrt(256)
      float s1 = S1[j] * 0.0625f;
      const int k0 = kb + ln, k1 = kb + 16 + ln;
      if (k0 > q_t || k0 > TSEQ) s0 = -1.0e30f;
      if (k1 > q_t || k1 > TSEQ) s1 = -1.0e30f;

      float mx = fmaxf(s0, s1);
      mx = fmaxf(mx, __shfl_xor(mx, 1));
      mx = fmaxf(mx, __shfl_xor(mx, 2));
      mx = fmaxf(mx, __shfl_xor(mx, 4));
      mx = fmaxf(mx, __shfl_xor(mx, 8));
      const float mnew = fmaxf(m8[j], mx);

      const float p0 = __expf(s0 - mnew);
      const float p1 = __expf(s1 - mnew);
      float sum = p0 + p1;
      sum += __shfl_xor(sum, 1);
      sum += __shfl_xor(sum, 2);
      sum += __shfl_xor(sum, 4);
      sum += __shfl_xor(sum, 8);

      alpha[j] = __expf(m8[j] - mnew);
      l8[j] = l8[j] * alpha[j] + sum;
      m8[j] = mnew;

      const int prow = j + 8 * half;
      Pw[prow * 32 + ln]      = (bf16)p0;
      Pw[prow * 32 + 16 + ln] = (bf16)p1;
    }

    // ---- rescale running O ----
#pragma unroll
    for (int t = 0; t < 16; ++t)
#pragma unroll
      for (int j = 0; j < 8; ++j) O[t][j] *= alpha[j];

    // ---- O += P * V: A = P from LDS, B = Vt rows, pipelined ----
    v16bf pa   = frag_ld_contig(Pw + ln * 32, half);   // A: row = ln, K = 32 keys
    v16bf vb_c = frag_ld_contig(Vtb + (size_t)ln * VTS + kb, half);
#pragma unroll
    for (int t = 0; t < 16; ++t) {
      v16bf vb_n = vb_c;
      if (t < 15)
        vb_n = frag_ld_contig(Vtb + (size_t)((t + 1) * 16 + ln) * VTS + kb, half);
      O[t] = wmma_bf16(pa, vb_c, O[t]);
      vb_c = vb_n;
      SG(0x020, 2);   // 2 VMEM reads per WMMA
      SG(0x008, 1);
    }
  }

  // ---- epilogue: O/l, * sigmoid(g) (transposed b128 reads), store y ----
#pragma unroll
  for (int t = 0; t < 16; ++t) {
    const int d = t * 16 + ln;
    const v8bf g8 = *(const v8bf*)(Gtb + (size_t)d * VTS + qt0 + 8 * half);
#pragma unroll
    for (int j = 0; j < 8; ++j) {
      const int q_t = qt0 + j + 8 * half;
      if (q_t >= 1 && q_t <= TSEQ) {
        const float gv = (float)g8[j];
        const float o  = (O[t][j] / l8[j]) * (1.0f / (1.0f + __expf(-gv)));
        Y[((size_t)b * TSEQ + (q_t - 1)) * NQKV + h * HDIM + d] = (bf16)o;
      }
    }
  }
}

// ---------------------------------------------------------------------------
// Output projection: out = y @ W_out^T   (M=8192, K=2048, N=256) -> f32
// grid: (512, 2), block: 128 (4 waves, each owns 32 cols), pipelined K loop
// ---------------------------------------------------------------------------
__global__ void __launch_bounds__(128)
outproj_kernel(const bf16* __restrict__ Yb, const bf16* __restrict__ Wob,
               float* __restrict__ Out) {
  const int lane = threadIdx.x & 31;
  const int wave = threadIdx.x >> 5;
  const int half = lane >> 4;
  const int ln   = lane & 15;
  const int m0   = blockIdx.x * 16;
  const int nb   = blockIdx.y * 128 + wave * 32;

  v8f acc0 = zero8(), acc1 = zero8();
  const bf16* yr = Yb  + (size_t)(m0 + ln) * NQKV;
  const bf16* w0 = Wob + (size_t)(nb + ln) * NQKV;
  const bf16* w1 = Wob + (size_t)(nb + 16 + ln) * NQKV;

  v16bf a_c  = frag_ld_contig(yr, half);
  v16bf b0_c = frag_ld_contig(w0, half);
  v16bf b1_c = frag_ld_contig(w1, half);
  for (int kk = 0; kk < 64; ++kk) {     // K = 2048 = 64 x 32
    v16bf a_n = a_c, b0_n = b0_c, b1_n = b1_c;
    if (kk < 63) {
      a_n  = frag_ld_contig(yr + (kk + 1) * 32, half);
      b0_n = frag_ld_contig(w0 + (kk + 1) * 32, half);
      b1_n = frag_ld_contig(w1 + (kk + 1) * 32, half);
    }
    acc0 = wmma_bf16(a_c, b0_c, acc0);
    acc1 = wmma_bf16(a_c, b1_c, acc1);
    a_c = a_n; b0_c = b0_n; b1_c = b1_n;
    SG(0x020, 3);   // 6 VMEM reads / 2 WMMAs per kk, interleaved
    SG(0x008, 1);
    SG(0x020, 3);
    SG(0x008, 1);
  }

#pragma unroll
  for (int j = 0; j < 8; ++j) {
    const int row = m0 + j + 8 * half;
    Out[(size_t)row * CEMB + nb + ln]      = acc0[j];
    Out[(size_t)row * CEMB + nb + 16 + ln] = acc1[j];
  }
}

// ---------------------------------------------------------------------------
// Host-side orchestration
// ---------------------------------------------------------------------------
extern "C" void kernel_launch(void* const* d_in, const int* in_sizes, int n_in,
                              void* d_out, int out_size, void* d_ws, size_t ws_size,
                              hipStream_t stream) {
  const float* x      = (const float*)d_in[0];
  const float* cosp   = (const float*)d_in[1];
  const float* sinp   = (const float*)d_in[2];
  const float* W_qkvg = (const float*)d_in[3];
  const float* W_sink = (const float*)d_in[4];
  const float* W_out  = (const float*)d_in[5];
  const float* tao    = (const float*)d_in[6];
  (void)in_sizes; (void)n_in; (void)out_size; (void)ws_size;

  // workspace layout (all sizes 256B-aligned)
  char* ws = (char*)d_ws;
  size_t off = 0;
  bf16* x0b = (bf16*)(ws + off); off += (size_t)BATCH * T1 * CEMB * 2;       // 4.20 MB
  bf16* wqb = (bf16*)(ws + off); off += (size_t)4 * NQKV * CEMB * 2;         // 4.19 MB
  bf16* wob = (bf16*)(ws + off); off += (size_t)CEMB * NQKV * 2;             // 1.05 MB
  bf16* Qb  = (bf16*)(ws + off); off += (size_t)BATCH * NHEAD * T1 * HDIM * 2;
  bf16* Kb  = (bf16*)(ws + off); off += (size_t)BATCH * NHEAD * T1 * HDIM * 2;
  bf16* Vtb = (bf16*)(ws + off); off += (size_t)BATCH * NHEAD * HDIM * VTS * 2;  // 34.1 MB
  bf16* Gtb = (bf16*)(ws + off); off += (size_t)BATCH * NHEAD * HDIM * VTS * 2;  // 34.1 MB
  bf16* Yb  = (bf16*)(ws + off); off += (size_t)BATCH * TSEQ * NQKV * 2;     // total ~178 MB

  // 1) fp32 -> bf16 weight conversion, x0 assembly, Vt pad zero-init
  {
    int nwq = 4 * NQKV * CEMB;           // 2,097,152
    cvt_f32_bf16<<<(nwq + 255) / 256, 256, 0, stream>>>(W_qkvg, wqb, nwq);
    int nwo = CEMB * NQKV;               // 524,288
    cvt_f32_bf16<<<(nwo + 255) / 256, 256, 0, stream>>>(W_out, wob, nwo);
    int nx0 = BATCH * T1 * CEMB;         // 2,098,176
    build_x0<<<(nx0 + 255) / 256, 256, 0, stream>>>(x, W_sink, x0b);
    int nvt = BATCH * NHEAD * HDIM * VTS;  // 17,039,360
    zero_bf16<<<(nvt + 255) / 256, 256, 0, stream>>>(Vtb, nvt);
  }

  // 2) fused QKVG projection + RoPE + RMSNorm (WMMA GEMM per tile)
  {
    dim3 grid(129, BATCH * NHEAD, 4);
    qkvg_kernel<<<grid, 64, 0, stream>>>(x0b, wqb, cosp, sinp, tao, Qb, Kb, Vtb, Gtb);
  }

  // 3) flash attention with sigmoid(g) gating (WMMA for S and P*V)
  {
    dim3 grid(33, BATCH * NHEAD);   // 33*4 waves cover 129 q-tiles
    attn_kernel<<<grid, 128, 0, stream>>>(Qb, Kb, Vtb, Gtb, Yb);
  }

  // 4) output projection (WMMA GEMM) -> f32 d_out
  {
    dim3 grid((BATCH * TSEQ) / 16, 2);
    outproj_kernel<<<grid, 128, 0, stream>>>(Yb, wob, (float*)d_out);
  }
}
